// VanillaRNN_14628658610559
// MI455X (gfx1250) — matrix operand order repaired and checked by reading, not compiled, not run on hardware
//
#include <hip/hip_runtime.h>
#include <math.h>

#define SEQ_LEN     256
#define INPUT_DIM   256
#define NUM_HIDDEN  1024
#define NUM_CLASSES 1024
#define BATCH       256

#define BM 128
#define BN 128
#define BK 32

typedef __attribute__((ext_vector_type(16))) __bf16 v16bf;
typedef __attribute__((ext_vector_type(8)))  __bf16 v8bf;
typedef __attribute__((ext_vector_type(8)))  float  v8f;

// ---------------------------------------------------------------- helpers

__device__ __forceinline__ unsigned short f2bf(float f) {
    __bf16 b = (__bf16)f;
    return __builtin_bit_cast(unsigned short, b);
}

#if __has_builtin(__builtin_amdgcn_tanhf)
#define FAST_TANH(x) __builtin_amdgcn_tanhf(x)
#elif __has_builtin(__builtin_amdgcn_tanh_f32)
#define FAST_TANH(x) __builtin_amdgcn_tanh_f32(x)
#else
#define FAST_TANH(x) tanhf(x)
#endif

// A-matrix 16x32 bf16 fragment (ISA 7.12.2 layout):
// lane<16 : row M=lane,    VGPR0-3 hold K=0..7,  VGPR4-7 hold K=16..23
// lane>=16: row M=lane-16, VGPR0-3 hold K=8..15, VGPR4-7 hold K=24..31
__device__ __forceinline__ v16bf load_fragA(const unsigned short* tile, int lane) {
    int half = lane >> 4;
    int r    = lane & 15;
    const v8bf* lo = (const v8bf*)(tile + r * BK + half * 8);
    const v8bf* hi = (const v8bf*)(tile + r * BK + half * 8 + 16);
    return __builtin_shufflevector(*lo, *hi,
        0, 1, 2, 3, 4, 5, 6, 7, 8, 9, 10, 11, 12, 13, 14, 15);
}

// B-matrix 32x16 bf16 fragment: lane holds column N=lane%16, same K order as A
// (lanes 0-15 -> K=0..7,16..23, lanes 16-31 -> K=8..15,24..31).
// B tile lives in LDS as [N][BK]: two 16-byte reads.
__device__ __forceinline__ v16bf load_fragB(const unsigned short* tile, int lane) {
    int half = lane >> 4;
    int c    = lane & 15;
    const v8bf* lo = (const v8bf*)(tile + c * BK + half * 8);
    const v8bf* hi = (const v8bf*)(tile + c * BK + half * 8 + 16);
    return __builtin_shufflevector(*lo, *hi,
        0, 1, 2, 3, 4, 5, 6, 7, 8, 9, 10, 11, 12, 13, 14, 15);
}

__device__ __forceinline__ void mma_tiles(
    const unsigned short* sA, const unsigned short* sB,
    int wm, int wn, int lane, v8f acc[2][4])
{
    v16bf a[2], b[4];
#pragma unroll
    for (int mi = 0; mi < 2; ++mi)
        a[mi] = load_fragA(sA + (wm + mi * 16) * BK, lane);
#pragma unroll
    for (int ni = 0; ni < 4; ++ni)
        b[ni] = load_fragB(sB + (wn + ni * 16) * BK, lane);
#pragma unroll
    for (int mi = 0; mi < 2; ++mi)
#pragma unroll
        for (int ni = 0; ni < 4; ++ni)
            acc[mi][ni] = __builtin_amdgcn_wmma_f32_16x16x32_bf16(
                false, a[mi], false, b[ni], (short)0, acc[mi][ni], false, false);
}

// LDS byte offset of a __shared__ object: generic LDS addresses keep the
// wave-relative LDS offset in the low 32 bits.
__device__ __forceinline__ unsigned lds_off(const void* p) {
    return (unsigned)(unsigned long long)p;
}

__device__ __forceinline__ void wait_async0() {
    asm volatile("s_wait_asynccnt 0x0" ::: "memory");
}

// Stage one K-chunk with ASYNC global->LDS copies (no VGPR round trip,
// tracked on ASYNCcnt; INST_OFFSET applies to both global and LDS address,
// so each thread's 32 contiguous bytes = two b128 async ops).
template <bool HAS_X>
__device__ __forceinline__ void stage_chunk_async(
    const unsigned short* __restrict__ Ax,   // [BATCH][INPUT_DIM]
    const unsigned short* __restrict__ Wxt,  // [NUM_HIDDEN][INPUT_DIM]
    const unsigned short* __restrict__ Ah,   // [BATCH][NUM_HIDDEN]
    const unsigned short* __restrict__ Wht,  // [1024][NUM_HIDDEN]
    int c, int m0, int n0,
    unsigned short* sA, unsigned short* sB, int tid)
{
    const unsigned short* A;
    const unsigned short* Bt;
    int ld, k0;
    if (HAS_X && c < INPUT_DIM / BK) {
        A = Ax;  Bt = Wxt;  ld = INPUT_DIM;  k0 = c * BK;
    } else {
        A = Ah;  Bt = Wht;  ld = NUM_HIDDEN;
        k0 = (c - (HAS_X ? INPUT_DIM / BK : 0)) * BK;
    }
    const int row = tid >> 1;          // 0..127
    const int kh  = (tid & 1) * 16;    // 0 / 16

    unsigned long long gA =
        (unsigned long long)(A + (size_t)(m0 + row) * ld + k0 + kh);
    unsigned long long gB =
        (unsigned long long)(Bt + (size_t)(n0 + row) * ld + k0 + kh);
    unsigned lA = lds_off(sA + row * BK + kh);
    unsigned lB = lds_off(sB + row * BK + kh);

    asm volatile(
        "global_load_async_to_lds_b128 %0, %2, off\n\t"
        "global_load_async_to_lds_b128 %0, %2, off offset:16\n\t"
        "global_load_async_to_lds_b128 %1, %3, off\n\t"
        "global_load_async_to_lds_b128 %1, %3, off offset:16"
        :
        : "v"(lA), "v"(lB), "v"(gA), "v"(gB)
        : "memory");
}

// ---------------------------------------------------------------- GEMM step
// C[256,1024] = epilogue( [x_t @ W_hx] + h @ W_hh + bias )
template <bool HAS_X, bool DO_TANH>
__global__ __launch_bounds__(256) void rnn_gemm(
    const unsigned short* __restrict__ Ax,
    const unsigned short* __restrict__ Wxt,
    const unsigned short* __restrict__ Ah,
    const unsigned short* __restrict__ Wht,
    const float* __restrict__ bias,
    unsigned short* __restrict__ hout,       // bf16 out (DO_TANH)
    float* __restrict__ fout)                // fp32 out (!DO_TANH)
{
    __shared__ alignas(64) unsigned short sA[2][BM * BK];
    __shared__ alignas(64) unsigned short sB[2][BN * BK];

    const int tid  = threadIdx.x;
    const int lane = tid & 31;
    const int wave = tid >> 5;

    const int m0blk = blockIdx.y * BM;       // 0 / 128
    const int n0blk = blockIdx.x * BN;       // 0..896
    const int wm = (wave >> 1) * 32;
    const int wn = (wave & 1) * 64;

    v8f acc[2][4];
#pragma unroll
    for (int mi = 0; mi < 2; ++mi)
#pragma unroll
        for (int ni = 0; ni < 4; ++ni)
            acc[mi][ni] = (v8f){0.f, 0.f, 0.f, 0.f, 0.f, 0.f, 0.f, 0.f};

    const int NC = (HAS_X ? (INPUT_DIM + NUM_HIDDEN) : NUM_HIDDEN) / BK;

    // Software-pipelined, double-buffered main loop: async-stage chunk c+1
    // straight into LDS while the WMMAs for chunk c issue; the asynccnt
    // wait overlaps the whole compute block. One barrier per chunk.
    stage_chunk_async<HAS_X>(Ax, Wxt, Ah, Wht, 0, m0blk, n0blk,
                             sA[0], sB[0], tid);
    wait_async0();
    __syncthreads();
    for (int c = 0; c < NC; ++c) {
        const int cur = c & 1;
        if (c + 1 < NC)
            stage_chunk_async<HAS_X>(Ax, Wxt, Ah, Wht, c + 1, m0blk, n0blk,
                                     sA[cur ^ 1], sB[cur ^ 1], tid);
        mma_tiles(sA[cur], sB[cur], wm, wn, lane, acc);
        wait_async0();
        __syncthreads();
    }

    // Epilogue. C/D layout: VGPR j -> M = j + 8*(lane>=16), N = lane%16.
    const int half = lane >> 4;
    const int r    = lane & 15;
#pragma unroll
    for (int mi = 0; mi < 2; ++mi) {
#pragma unroll
        for (int ni = 0; ni < 4; ++ni) {
            const int col = n0blk + wn + ni * 16 + r;
            const float bv = bias[col];
#pragma unroll
            for (int j = 0; j < 8; ++j) {
                const int row = m0blk + wm + mi * 16 + half * 8 + j;
                const float v = acc[mi][ni][j] + bv;
                if (DO_TANH)
                    hout[(size_t)row * NUM_HIDDEN + col] = f2bf(FAST_TANH(v));
                else
                    fout[(size_t)row * NUM_CLASSES + col] = v;
            }
        }
    }
}

// ---------------------------------------------------------------- prep

__global__ void f32_to_bf16(const float* __restrict__ src,
                            unsigned short* __restrict__ dst, int n) {
    int i = blockIdx.x * blockDim.x + threadIdx.x;
    if (i < n) dst[i] = f2bf(src[i]);
}

// W[k][n] fp32 (row stride 1024) -> Wt[n][k] bf16 (row stride K).
// Paid once; keeps the per-step staging as pure b128 copies.
__global__ void transpose_w(const float* __restrict__ W,
                            unsigned short* __restrict__ Wt, int K) {
    __shared__ float tile[32][33];
    const int k0 = blockIdx.x * 32;
    const int n0 = blockIdx.y * 32;
    const int tx = threadIdx.x;   // 0..31
    const int ty = threadIdx.y;   // 0..7
#pragma unroll
    for (int i = ty; i < 32; i += 8)                       // tile[k][n]
        tile[i][tx] = W[(size_t)(k0 + i) * NUM_HIDDEN + n0 + tx];
    __syncthreads();
#pragma unroll
    for (int i = ty; i < 32; i += 8)                       // Wt[n][k], k coalesced
        Wt[(size_t)(n0 + i) * K + k0 + tx] = f2bf(tile[tx][i]);
}

// x[b][d][t] fp32 -> xt[t][b][d] bf16, LDS 32x32 tile transpose per b-plane.
__global__ void transpose_x(const float* __restrict__ x,
                            unsigned short* __restrict__ xt) {
    __shared__ float tile[32][33];
    const int t0 = blockIdx.x * 32;
    const int d0 = blockIdx.y * 32;
    const int b  = blockIdx.z;
    const int tx = threadIdx.x;
    const int ty = threadIdx.y;

    const float* src = x + (size_t)b * INPUT_DIM * SEQ_LEN
                         + (size_t)d0 * SEQ_LEN + t0;
#pragma unroll
    for (int i = ty; i < 32; i += 8)
        tile[i][tx] = src[(size_t)i * SEQ_LEN + tx];       // tile[d][t]
    __syncthreads();

    unsigned short* dst = xt + (size_t)t0 * BATCH * INPUT_DIM
                             + (size_t)b * INPUT_DIM + d0;
#pragma unroll
    for (int i = ty; i < 32; i += 8)                       // row t=t0+i, col d=d0+tx
        dst[(size_t)i * BATCH * INPUT_DIM + tx] = f2bf(tile[tx][i]);
}

// ---------------------------------------------------------------- launch

extern "C" void kernel_launch(void* const* d_in, const int* in_sizes, int n_in,
                              void* d_out, int out_size, void* d_ws, size_t ws_size,
                              hipStream_t stream) {
    (void)in_sizes; (void)n_in; (void)out_size; (void)ws_size;

    const float* x   = (const float*)d_in[0];
    const float* Whx = (const float*)d_in[1];
    const float* Whh = (const float*)d_in[2];
    const float* Wph = (const float*)d_in[3];
    const float* bh  = (const float*)d_in[4];
    const float* bp  = (const float*)d_in[5];
    const float* h0  = (const float*)d_in[6];
    float* out = (float*)d_out;

    // Workspace layout (bf16 stored as raw u16), ~37.5 MB total.
    char* p = (char*)d_ws;
    unsigned short* xt  = (unsigned short*)p; p += (size_t)SEQ_LEN * BATCH * INPUT_DIM * 2;
    unsigned short* wxt = (unsigned short*)p; p += (size_t)NUM_HIDDEN * INPUT_DIM * 2;
    unsigned short* wht = (unsigned short*)p; p += (size_t)NUM_HIDDEN * NUM_HIDDEN * 2;
    unsigned short* wpt = (unsigned short*)p; p += (size_t)NUM_CLASSES * NUM_HIDDEN * 2;
    unsigned short* hA  = (unsigned short*)p; p += (size_t)BATCH * NUM_HIDDEN * 2;
    unsigned short* hB  = (unsigned short*)p; p += (size_t)BATCH * NUM_HIDDEN * 2;

    const int thr = 256;
    transpose_w<<<dim3(INPUT_DIM / 32, NUM_HIDDEN / 32), dim3(32, 8), 0, stream>>>(
        Whx, wxt, INPUT_DIM);
    transpose_w<<<dim3(NUM_HIDDEN / 32, NUM_HIDDEN / 32), dim3(32, 8), 0, stream>>>(
        Whh, wht, NUM_HIDDEN);
    transpose_w<<<dim3(NUM_HIDDEN / 32, NUM_CLASSES / 32), dim3(32, 8), 0, stream>>>(
        Wph, wpt, NUM_HIDDEN);
    f32_to_bf16<<<(BATCH * NUM_HIDDEN + thr - 1) / thr, thr, 0, stream>>>(
        h0, hA, BATCH * NUM_HIDDEN);
    transpose_x<<<dim3(SEQ_LEN / 32, INPUT_DIM / 32, BATCH), dim3(32, 8), 0, stream>>>(
        x, xt);

    dim3 grid(NUM_HIDDEN / BN, BATCH / BM);   // (8, 2) blocks of 256 threads

    unsigned short* hcur = hA;
    unsigned short* hnext = hB;
    for (int t = 0; t < SEQ_LEN; ++t) {
        rnn_gemm<true, true><<<grid, 256, 0, stream>>>(
            xt + (size_t)t * BATCH * INPUT_DIM, wxt, hcur, wht, bh, hnext, nullptr);
        unsigned short* tmp = hcur; hcur = hnext; hnext = tmp;
    }

    rnn_gemm<false, false><<<grid, 256, 0, stream>>>(
        nullptr, nullptr, hcur, wpt, bp, nullptr, out);
}
